// CausalSelfAttention_36825049596521
// MI455X (gfx1250) — matrix operand order repaired and checked by reading, not compiled, not run on hardware
//
#include <hip/hip_runtime.h>

// ---------------------------------------------------------------------------
// CausalSelfAttention forward for MI455X (gfx1250, wave32, WMMA bf16,
// async global->LDS copies). B=8, T=1024, C=2048, H=16, HD=128.
// All intermediates kept in bf16 (qkv 96MB, attn 32MB) to halve HBM traffic.
// ---------------------------------------------------------------------------

#define Bsz 8
#define Tsz 1024
#define Csz 2048
#define Hsz 16
#define HDsz 128
#define QKV_LD (3 * Csz)            // 6144
#define SCALE 0.08838834764831845f  // 1/sqrt(128), folded into q in GEMM1

typedef __bf16 bf16_t;
typedef bf16_t v16bf __attribute__((ext_vector_type(16)));
typedef bf16_t v8bf  __attribute__((ext_vector_type(8)));
typedef bf16_t v4bf  __attribute__((ext_vector_type(4)));
typedef float  v8f   __attribute__((ext_vector_type(8)));

__device__ __forceinline__ v16bf comb(v8bf lo, v8bf hi) {
    v16bf r;
#pragma unroll
    for (int i = 0; i < 8; ++i) { r[i] = lo[i]; r[i + 8] = hi[i]; }
    return r;
}

// A-fragment (16x32 bf16): lane row = L&15; chunks at k0 and k0+16 (k0=half*8).
__device__ __forceinline__ v16bf ldA(const bf16_t* row, int k0) {
    return comb(*(const v8bf*)(row + k0), *(const v8bf*)(row + k0 + 16));
}
// B-fragment from transposed tile Bt[n][k]: chunks at k0 and k0+16 (k0=half*8).
__device__ __forceinline__ v16bf ldB(const bf16_t* row, int k0) {
    return comb(*(const v8bf*)(row + k0), *(const v8bf*)(row + k0 + 16));
}

__device__ __forceinline__ v8f wmma_bf(v16bf a, v16bf b, v8f c) {
    return __builtin_amdgcn_wmma_f32_16x16x32_bf16(
        false, a, false, b, (short)0, c, false, false);
}

// Async 16-byte global -> LDS copy (CDNA5 GLOBAL_LOAD_ASYNC_TO_LDS_B128, GV).
__device__ __forceinline__ void async_g2l_b128(unsigned lds_addr, const void* gptr) {
    unsigned long long ga = (unsigned long long)gptr;
    asm volatile("global_load_async_to_lds_b128 %0, %1, off"
                 :: "v"(lds_addr), "v"(ga) : "memory");
}
__device__ __forceinline__ void wait_async0() {
    asm volatile("s_wait_asynccnt 0" ::: "memory");
}

// ---------------------------------------------------------------------------
// GEMM: out[M,N] = A[M,K] * W[K,N] (f32) + bias, tile 64x64, K-step 64.
// 4 waves, each 32x32 (2x2 WMMA tiles). A is f32 (convert) or bf16 (async DMA).
// Output f32 or bf16; q-scale folded for col < scaleCols.
// ---------------------------------------------------------------------------
template <bool ABF, bool OUTBF>
__global__ __launch_bounds__(128)
void gemm_kernel(const void* __restrict__ Ap, const float* __restrict__ W,
                 const float* __restrict__ bias, void* __restrict__ outp,
                 int M, int K, int N, int scaleCols) {
    __shared__ __attribute__((aligned(16))) bf16_t As[64][72];   // [m][k], pad 8
    __shared__ __attribute__((aligned(16))) bf16_t Bs[64][72];   // [n][k] transposed

    const int tid  = threadIdx.x;
    const int lane = tid & 31;
    const int wave = tid >> 5;
    const int half = lane >> 4;
    const int l15  = lane & 15;

    const int mb = blockIdx.y * 64;
    const int nb = blockIdx.x * 64;
    const int wm = (wave >> 1) * 32;
    const int wn = (wave & 1) * 32;

    v8f acc[2][2] = {};

    for (int kk = 0; kk < K; kk += 64) {
        if (ABF) {
            // A tile 64x64 bf16 = 512 x 16B chunks: async DMA, 4 per thread
            const bf16_t* A = (const bf16_t*)Ap;
#pragma unroll
            for (int p = 0; p < 4; ++p) {
                int idx = tid + p * 128;          // 0..511
                int r  = idx >> 3;                // row 0..63
                int c8 = idx & 7;                 // 8-elem chunk 0..7
                async_g2l_b128((unsigned)(size_t)&As[r][c8 * 8],
                               A + (size_t)(mb + r) * K + kk + c8 * 8);
            }
        } else {
            // A tile 64x64 f32 -> bf16 via hardware cvt, 8 float4 per thread
            const float* A = (const float*)Ap;
#pragma unroll
            for (int p = 0; p < 8; ++p) {
                int idx = tid + p * 128;          // 0..1023
                int r  = idx >> 4;                // row 0..63
                int c4 = idx & 15;                // float4 col 0..15
                float4 v = *(const float4*)(A + (size_t)(mb + r) * K + kk + c4 * 4);
                v4bf t;
                t[0] = (bf16_t)v.x; t[1] = (bf16_t)v.y;
                t[2] = (bf16_t)v.z; t[3] = (bf16_t)v.w;
                *(v4bf*)&As[r][c4 * 4] = t;
            }
        }
        // W tile 64x64 f32, transposed into Bs[n][k], 8 float4 per thread
#pragma unroll
        for (int p = 0; p < 8; ++p) {
            int idx = tid + p * 128;
            int r  = idx >> 4;                    // k row 0..63
            int c4 = idx & 15;                    // float4 col (n) 0..15
            float4 v = *(const float4*)(W + (size_t)(kk + r) * N + nb + c4 * 4);
            Bs[c4 * 4 + 0][r] = (bf16_t)v.x;
            Bs[c4 * 4 + 1][r] = (bf16_t)v.y;
            Bs[c4 * 4 + 2][r] = (bf16_t)v.z;
            Bs[c4 * 4 + 3][r] = (bf16_t)v.w;
        }
        if (ABF) wait_async0();
        __syncthreads();

#pragma unroll
        for (int ks = 0; ks < 2; ++ks) {
            v16bf af[2], bw[2];
#pragma unroll
            for (int mt = 0; mt < 2; ++mt)
                af[mt] = ldA(&As[wm + mt * 16 + l15][0], ks * 32 + half * 8);
#pragma unroll
            for (int nt = 0; nt < 2; ++nt)
                bw[nt] = ldB(&Bs[wn + nt * 16 + l15][0], ks * 32 + half * 8);
#pragma unroll
            for (int mt = 0; mt < 2; ++mt)
#pragma unroll
                for (int nt = 0; nt < 2; ++nt)
                    acc[mt][nt] = wmma_bf(af[mt], bw[nt], acc[mt][nt]);
        }
        __syncthreads();
    }

    // Epilogue (C/D layout): bias add, optional q-scale, f32 or bf16 store
#pragma unroll
    for (int nt = 0; nt < 2; ++nt) {
        int col = nb + wn + nt * 16 + l15;
        float bv  = bias[col];
        float scl = (col < scaleCols) ? SCALE : 1.0f;
#pragma unroll
        for (int mt = 0; mt < 2; ++mt) {
            int rb = mb + wm + mt * 16 + half * 8;
#pragma unroll
            for (int i = 0; i < 8; ++i) {
                float val = (acc[mt][nt][i] + bv) * scl;
                if (OUTBF)
                    ((bf16_t*)outp)[(size_t)(rb + i) * N + col] = (bf16_t)val;
                else
                    ((float*)outp)[(size_t)(rb + i) * N + col] = val;
            }
        }
    }
}

// ---------------------------------------------------------------------------
// Flash attention: workgroup = (b, h, 64 q-rows); 4 waves x 16 q-rows.
// qkv is bf16 [B,T,6144]; q pre-scaled by 1/sqrt(HD) in GEMM1.
// K tile staged via async DMA; V staged transposed; output bf16.
// ---------------------------------------------------------------------------
__global__ __launch_bounds__(128)
void attention_kernel(const bf16_t* __restrict__ qkv, bf16_t* __restrict__ attn) {
    __shared__ __attribute__((aligned(16))) bf16_t Kt[64][136];     // [key][d]
    __shared__ __attribute__((aligned(16))) bf16_t Vt[128][72];     // [d][key]
    __shared__ __attribute__((aligned(16))) bf16_t Pl[4][16][72];   // per-wave P

    const int tid  = threadIdx.x;
    const int lane = tid & 31;
    const int wave = tid >> 5;
    const int half = lane >> 4;
    const int l15  = lane & 15;

    const int qb = blockIdx.x;
    const int bi = blockIdx.y >> 4;
    const int h  = blockIdx.y & 15;
    const int qrow0 = qb * 64 + wave * 16;

    // Q fragments: pure 16B vector loads (already scaled)
    v16bf qf[4];
    {
        const bf16_t* qp = qkv + (size_t)(bi * Tsz + qrow0 + l15) * QKV_LD + h * HDsz;
#pragma unroll
        for (int ds = 0; ds < 4; ++ds) {
            int k0 = ds * 32 + half * 8;
            qf[ds] = comb(*(const v8bf*)(qp + k0), *(const v8bf*)(qp + k0 + 16));
        }
    }

    v8f o[8];
#pragma unroll
    for (int nt = 0; nt < 8; ++nt) o[nt] = (v8f){};
    float row_max[8], row_sum[8];
#pragma unroll
    for (int i = 0; i < 8; ++i) { row_max[i] = -1e30f; row_sum[i] = 0.f; }

    for (int kb = 0; kb <= qb; ++kb) {
        const bf16_t* kp = qkv + (size_t)(bi * Tsz + kb * 64) * QKV_LD + Csz + h * HDsz;
        const bf16_t* vp = qkv + (size_t)(bi * Tsz + kb * 64) * QKV_LD + 2 * Csz + h * HDsz;

        // K tile [64][128] bf16 = 1024 x 16B chunks: async DMA, 8 per thread
#pragma unroll
        for (int p = 0; p < 8; ++p) {
            int idx = tid + p * 128;          // 0..1023
            int r   = idx >> 4;               // key 0..63
            int c16 = idx & 15;               // 8-elem chunk 0..15
            async_g2l_b128((unsigned)(size_t)&Kt[r][c16 * 8],
                           kp + (size_t)r * QKV_LD + c16 * 8);
        }
        // V tile transposed into Vt[d][key] (vector load + scatter stores)
#pragma unroll
        for (int p = 0; p < 8; ++p) {
            int idx = tid + p * 128;
            int r   = idx >> 4;               // key 0..63
            int c16 = idx & 15;               // 8-elem chunk along d
            v8bf v = *(const v8bf*)(vp + (size_t)r * QKV_LD + c16 * 8);
#pragma unroll
            for (int j = 0; j < 8; ++j)
                Vt[c16 * 8 + j][r] = v[j];
        }
        wait_async0();
        __syncthreads();

        // S = Q K^T : 4 N-tiles x 4 d-steps
        v8f s[4];
#pragma unroll
        for (int nt = 0; nt < 4; ++nt) {
            v8f a = (v8f){};
#pragma unroll
            for (int ds = 0; ds < 4; ++ds) {
                v16bf bfk = ldB(&Kt[nt * 16 + l15][0], ds * 32 + half * 8);
                a = wmma_bf(qf[ds], bfk, a);
            }
            s[nt] = a;
        }

        // Causal mask: diagonal block only
        if (kb == qb) {
#pragma unroll
            for (int nt = 0; nt < 4; ++nt) {
                int col = kb * 64 + nt * 16 + l15;
#pragma unroll
                for (int i = 0; i < 8; ++i) {
                    int row = qrow0 + half * 8 + i;
                    if (col > row) s[nt][i] = -1e30f;
                }
            }
        }

        // Online softmax (reductions across each 16-lane half)
#pragma unroll
        for (int i = 0; i < 8; ++i) {
            float m = fmaxf(fmaxf(s[0][i], s[1][i]), fmaxf(s[2][i], s[3][i]));
            m = fmaxf(m, __shfl_xor(m, 1, 32));
            m = fmaxf(m, __shfl_xor(m, 2, 32));
            m = fmaxf(m, __shfl_xor(m, 4, 32));
            m = fmaxf(m, __shfl_xor(m, 8, 32));
            float nm = fmaxf(row_max[i], m);
            float alpha = __expf(row_max[i] - nm);
            row_max[i] = nm;
            float rs = 0.f;
#pragma unroll
            for (int nt = 0; nt < 4; ++nt) {
                float p = __expf(s[nt][i] - nm);
                s[nt][i] = p;
                rs += p;
            }
            rs += __shfl_xor(rs, 1, 32);
            rs += __shfl_xor(rs, 2, 32);
            rs += __shfl_xor(rs, 4, 32);
            rs += __shfl_xor(rs, 8, 32);
            row_sum[i] = row_sum[i] * alpha + rs;
#pragma unroll
            for (int nt = 0; nt < 8; ++nt) o[nt][i] *= alpha;
        }

        // P: C/D layout -> A layout via per-wave LDS (same-wave LDS in order)
#pragma unroll
        for (int nt = 0; nt < 4; ++nt)
#pragma unroll
            for (int i = 0; i < 8; ++i)
                Pl[wave][half * 8 + i][nt * 16 + l15] = (bf16_t)s[nt][i];

        // O += P V : 2 k-steps x 8 d-tiles
#pragma unroll
        for (int ks = 0; ks < 2; ++ks) {
            v16bf af = ldA(&Pl[wave][l15][0], ks * 32 + half * 8);
#pragma unroll
            for (int nt = 0; nt < 8; ++nt) {
                v16bf bfv = ldB(&Vt[nt * 16 + l15][0], ks * 32 + half * 8);
                o[nt] = wmma_bf(af, bfv, o[nt]);
            }
        }
        __syncthreads();
    }

    // Normalize, write bf16 [B,T,C] (head h -> cols h*128..+127)
#pragma unroll
    for (int i = 0; i < 8; ++i) {
        float inv = 1.0f / row_sum[i];
        int row = qrow0 + half * 8 + i;
        bf16_t* orow = attn + (size_t)(bi * Tsz + row) * Csz + h * HDsz + l15;
#pragma unroll
        for (int nt = 0; nt < 8; ++nt)
            orow[nt * 16] = (bf16_t)(o[nt][i] * inv);
    }
}

// ---------------------------------------------------------------------------
extern "C" void kernel_launch(void* const* d_in, const int* in_sizes, int n_in,
                              void* d_out, int out_size, void* d_ws, size_t ws_size,
                              hipStream_t stream) {
    const float* x      = (const float*)d_in[0];   // [B,T,C]
    const float* w_attn = (const float*)d_in[1];   // [C,3C]
    const float* b_attn = (const float*)d_in[2];   // [3C]
    const float* w_proj = (const float*)d_in[3];   // [C,C]
    const float* b_proj = (const float*)d_in[4];   // [C]

    const size_t M = (size_t)Bsz * Tsz;                        // 8192
    const size_t QKV_BYTES = M * QKV_LD * sizeof(bf16_t);      // 96 MB
    bf16_t* qkv  = (bf16_t*)d_ws;
    bf16_t* attn = (bf16_t*)((char*)d_ws + QKV_BYTES);         // 32 MB

    // 1) qkv = x @ w_attn + b_attn  (bf16 out, q columns pre-scaled)
    gemm_kernel<false, true><<<dim3(QKV_LD / 64, M / 64), 128, 0, stream>>>(
        x, w_attn, b_attn, qkv, (int)M, Csz, QKV_LD, Csz);

    // 2) flash attention per (b,h,q-block), bf16 in/out
    attention_kernel<<<dim3(Tsz / 64, Bsz * Hsz), 128, 0, stream>>>(qkv, attn);

    // 3) out = attn @ w_proj + b_proj  (bf16 A via async DMA, f32 out)
    gemm_kernel<true, false><<<dim3(Csz / 64, M / 64), 128, 0, stream>>>(
        attn, w_proj, b_proj, d_out, (int)M, Csz, Csz, 0);
}
